// MyMSA_6184752906579
// MI455X (gfx1250) — matrix operand order repaired and checked by reading, not compiled, not run on hardware
//
#include <hip/hip_runtime.h>
#include <hip/hip_bf16.h>

// Problem sizes (match reference)
#define B_  4
#define S_  2048
#define D_  1024
#define H_  16
#define DH_ 64

typedef __attribute__((ext_vector_type(16))) _Float16 v16h;
typedef __attribute__((ext_vector_type(8)))  float    v8f;

union HV { v16h v; _Float16 h[16]; };
union FV { v8f  v; float    f[8];  };
union PK { uint4 u; _Float16 h[8]; };

// log2(e) / sqrt(DH)  -> softmax computed in base-2
#define CSCALE 0.1803368801111204f

// ---------------------------------------------------------------------------
// Kernel 0: bulk f32 -> f16 conversion (x and weights), 8 elems/thread.
// n must be a multiple of 2048 (holds for all our launches).
// ---------------------------------------------------------------------------
__global__ __launch_bounds__(256) void f32_to_f16_kernel(
    const float* __restrict__ src, _Float16* __restrict__ dst)
{
    const size_t i = ((size_t)blockIdx.x * 256 + threadIdx.x) * 8;
    float4 a = *(const float4*)(src + i);
    float4 b = *(const float4*)(src + i + 4);
    PK pk;
    pk.h[0] = (_Float16)a.x; pk.h[1] = (_Float16)a.y;
    pk.h[2] = (_Float16)a.z; pk.h[3] = (_Float16)a.w;
    pk.h[4] = (_Float16)b.x; pk.h[5] = (_Float16)b.y;
    pk.h[6] = (_Float16)b.z; pk.h[7] = (_Float16)b.w;
    *(uint4*)(dst + i) = pk.u;
}

// ---------------------------------------------------------------------------
// Kernel 1: per-head QKV projection.  One wave = 16 tokens x one head.
// All operands pre-converted f16 -> fragments are pure vector loads.
//
// Q,K  : A = W tile (16e x 32d), B = x^T (32d x 16tok) -> C rows = e.
//        A lane's 8 C registers are 8 consecutive e -> one b128 store per
//        tile into Q/K [B,H,S,DH].
// V    : A = x (16tok x 32d), B = W^T (32d x 16e)     -> C rows = token.
//        Stored transposed into V^T [B,H,DH,S]: one b128 store per tile.
// ---------------------------------------------------------------------------
__global__ __launch_bounds__(256) void qkv_proj_kernel(
    const _Float16* __restrict__ xh,
    const _Float16* __restrict__ Wqh, const _Float16* __restrict__ Wkh,
    const _Float16* __restrict__ Wvh,
    const float* __restrict__ bq, const float* __restrict__ bk,
    const float* __restrict__ bv,
    _Float16* __restrict__ Qo, _Float16* __restrict__ Ko,
    _Float16* __restrict__ Vo)
{
    const int lane = threadIdx.x & 31;
    const int wgid = blockIdx.x * 8 + (threadIdx.x >> 5);   // 0..8191
    const int h    = wgid & (H_ - 1);
    const int tb   = wgid >> 4;                             // 0..511
    const int tok0 = tb * 16;
    const int b    = tok0 / S_;
    const int s0   = tok0 % S_;

    const int M    = lane & 15;
    const int hi   = lane >> 4;       // half-wave select
    const int offA = hi ? 8 : 0;      // A-layout K offset within 32-chunk
    const int offB = hi ? 16 : 0;     // unused; B fragments also load at offA

    // x fragments (token row for this lane serves both A and B forms)
    const _Float16* xrow = xh + (size_t)(tok0 + M) * D_ + h * DH_;
    HV xa[2];   // A-form (V path): 16 tok x 32 d
    v16h xb[2]; // B-form (Q/K path): 32 d x 16 tok
#pragma unroll
    for (int kc = 0; kc < 2; ++kc) {
        const _Float16* p = xrow + kc * 32;
        *(uint4*)&xa[kc].h[0] = *(const uint4*)(p + offA);
        *(uint4*)&xa[kc].h[8] = *(const uint4*)(p + offA + 16);
        ((uint4*)&xb[kc])[0] = *(const uint4*)(p + offA);
        ((uint4*)&xb[kc])[1] = *(const uint4*)(p + offA + 16);
    }

    // ---- Q and K: transposed formulation ---------------------------------
    {
        const _Float16* Ws[2] = {Wqh, Wkh};
        const float*    Bs[2] = {bq, bk};
        _Float16*       Os[2] = {Qo, Ko};
#pragma unroll
        for (int m = 0; m < 2; ++m) {
            const _Float16* W    = Ws[m] + (size_t)h * DH_ * DH_;
            const float*    bias = Bs[m] + h * DH_;
            _Float16*       O    = Os[m] + ((size_t)(b * H_ + h) * S_ + s0) * DH_;
#pragma unroll
            for (int t = 0; t < 4; ++t) {
                const _Float16* wr = W + (size_t)(t * 16 + M) * DH_;  // e row
                HV wa0, wa1;
                *(uint4*)&wa0.h[0] = *(const uint4*)(wr + offA);
                *(uint4*)&wa0.h[8] = *(const uint4*)(wr + offA + 16);
                *(uint4*)&wa1.h[0] = *(const uint4*)(wr + 32 + offA);
                *(uint4*)&wa1.h[8] = *(const uint4*)(wr + 32 + offA + 16);
                v8f c = {};
                c = __builtin_amdgcn_wmma_f32_16x16x32_f16(false, wa0.v, false, xb[0],
                                                           (short)0, c, false, false);
                c = __builtin_amdgcn_wmma_f32_16x16x32_f16(false, wa1.v, false, xb[1],
                                                           (short)0, c, false, false);
                FV cf; cf.v = c;
                const float* bp = bias + t * 16 + hi * 8;   // 8 consecutive e
                PK pk;
#pragma unroll
                for (int r = 0; r < 8; ++r)
                    pk.h[r] = (_Float16)(cf.f[r] + bp[r]);
                // token = s0+M row; e = t*16 + hi*8 .. +7 contiguous
                *(uint4*)(O + (size_t)M * DH_ + t * 16 + hi * 8) = pk.u;
            }
        }
    }

    // ---- V: original formulation, stored transposed ----------------------
    {
        const _Float16* W    = Wvh + (size_t)h * DH_ * DH_;
        const float*    bias = bv + h * DH_;
        _Float16*       OV   = Vo + (size_t)(b * H_ + h) * DH_ * S_;   // V^T
#pragma unroll
        for (int t = 0; t < 4; ++t) {
            const _Float16* wr = W + (size_t)(t * 16 + M) * DH_;  // e = col n
            v16h wb0, wb1;
            ((uint4*)&wb0)[0] = *(const uint4*)(wr + offA);
            ((uint4*)&wb0)[1] = *(const uint4*)(wr + offA + 16);
            ((uint4*)&wb1)[0] = *(const uint4*)(wr + 32 + offA);
            ((uint4*)&wb1)[1] = *(const uint4*)(wr + 32 + offA + 16);
            v8f c = {};
            c = __builtin_amdgcn_wmma_f32_16x16x32_f16(false, xa[0].v, false, wb0,
                                                       (short)0, c, false, false);
            c = __builtin_amdgcn_wmma_f32_16x16x32_f16(false, xa[1].v, false, wb1,
                                                       (short)0, c, false, false);
            FV cf; cf.v = c;
            const float bb = bias[t * 16 + M];
            PK pk;
#pragma unroll
            for (int r = 0; r < 8; ++r)
                pk.h[r] = (_Float16)(cf.f[r] + bb);
            // V^T[e = t*16+M][tokens s0+hi*8 .. +7] contiguous
            *(uint4*)(OV + (size_t)(t * 16 + M) * S_ + s0 + hi * 8) = pk.u;
        }
    }
}

// ---------------------------------------------------------------------------
// Kernel 2: flash attention.  WG = 256 threads (8 waves) = 256 queries of one
// (b,h).  Each wave: 32 queries (two 16-row blocks).  Key loop: 32 keys/iter.
// K staged [key][dh], V staged transposed [dh][key] via DOUBLE-BUFFERED async
// LDS copies: next tile's global_load_async_to_lds_b128 is in flight while
// the current tile is consumed; s_wait_asynccnt 2 retires only the older pair.
// ---------------------------------------------------------------------------
#define QBLK 256
#define KT   32
#define TILE_HALF (KT * DH_)          // 2048 halves = 4 KB

__global__ __launch_bounds__(256) void attn_kernel(
    const _Float16* __restrict__ Q, const _Float16* __restrict__ K,
    const _Float16* __restrict__ Vt, float* __restrict__ out)
{
    __shared__ __align__(32) _Float16 Klds[2][TILE_HALF];    // 2 x 4 KB [key][dh]
    __shared__ __align__(32) _Float16 Vlds[2][TILE_HALF];    // 2 x 4 KB [dh][key]
    __shared__ __align__(32) _Float16 Plds[8][32 * KT];      // 8 x 2 KB

    const int tid  = threadIdx.x;
    const int lane = tid & 31;
    const int wv   = tid >> 5;

    const int nqb  = S_ / QBLK;                  // 8
    const int qblk = blockIdx.x % nqb;
    const int bh   = blockIdx.x / nqb;
    const int b    = bh / H_, h = bh % H_;

    const int q0 = qblk * QBLK + wv * 32;
    const _Float16* Qp = Q  + (size_t)(b * H_ + h) * S_ * DH_;
    const _Float16* Kp = K  + (size_t)(b * H_ + h) * S_ * DH_;
    const _Float16* Vp = Vt + (size_t)(b * H_ + h) * DH_ * S_;   // [dh][S]

    const int M    = lane & 15;
    const int hi   = lane >> 4;
    const int offA = hi ? 8 : 0;
    const int offB = hi ? 16 : 0;

    // LDS byte offsets + per-thread global source layout for async staging
    const unsigned ldsK0 = (unsigned)(uintptr_t)&Klds[0][0] + tid * 16;
    const unsigned ldsV0 = (unsigned)(uintptr_t)&Vlds[0][0] + tid * 16;
    const int vrow = tid >> 2, vch = tid & 3;    // V^T: 64 rows x 4 16B chunks

    auto issue_stage = [&](int kb, int bufidx) {
        unsigned long long gk =
            (unsigned long long)(uintptr_t)(Kp + (size_t)kb * DH_) + tid * 16;
        unsigned long long gv =
            (unsigned long long)(uintptr_t)(Vp + (size_t)vrow * S_ + kb) + vch * 16;
        unsigned lk = ldsK0 + bufidx * (TILE_HALF * 2);
        unsigned lv = ldsV0 + bufidx * (TILE_HALF * 2);
        asm volatile("global_load_async_to_lds_b128 %0, %1, off"
                     :: "v"(lk), "v"(gk) : "memory");
        asm volatile("global_load_async_to_lds_b128 %0, %1, off"
                     :: "v"(lv), "v"(gv) : "memory");
    };

    // --- Q fragments: 32 queries x 64 dh -> qa[rowblock][kchunk] ----------
    HV qa[2][2];
#pragma unroll
    for (int rb = 0; rb < 2; ++rb) {
        const _Float16* qrow = Qp + (size_t)(q0 + rb * 16 + M) * DH_;
#pragma unroll
        for (int kc = 0; kc < 2; ++kc) {
            *(uint4*)&qa[rb][kc].h[0] = *(const uint4*)(qrow + kc * 32 + offA);
            *(uint4*)&qa[rb][kc].h[8] = *(const uint4*)(qrow + kc * 32 + offA + 16);
        }
    }

    v8f o[2][4];
#pragma unroll
    for (int rb = 0; rb < 2; ++rb)
#pragma unroll
        for (int t = 0; t < 4; ++t) o[rb][t] = (v8f){};

    float mrow[2][8], lrow[2][8];
#pragma unroll
    for (int rb = 0; rb < 2; ++rb)
#pragma unroll
        for (int r = 0; r < 8; ++r) { mrow[rb][r] = -3.0e38f; lrow[rb][r] = 0.0f; }

    issue_stage(0, 0);                 // preload first tile

    for (int kb = 0; kb < S_; kb += KT) {
        const int  buf  = (kb >> 5) & 1;
        const bool more = (kb + KT) < S_;

        // prefetch next tile into the other buffer, then retire current tile
        if (more) {
            issue_stage(kb + KT, buf ^ 1);
            asm volatile("s_wait_asynccnt 2" ::: "memory");
        } else {
            asm volatile("s_wait_asynccnt 0" ::: "memory");
        }
        __syncthreads();

        const _Float16* Kt = &Klds[buf][0];
        const _Float16* Vb = &Vlds[buf][0];

        // ---- K B-fragments (dh x key), two 16B LDS reads at offA, offA+16 --
        v16h kf[2][2];
#pragma unroll
        for (int kt = 0; kt < 2; ++kt)
#pragma unroll
            for (int kc = 0; kc < 2; ++kc)
            {
                ((uint4*)&kf[kt][kc])[0] = *(const uint4*)(Kt + (kt * 16 + M) * DH_ + kc * 32 + offA);
                ((uint4*)&kf[kt][kc])[1] = *(const uint4*)(Kt + (kt * 16 + M) * DH_ + kc * 32 + offA + 16);
            }

        // ---- V B-fragments (key x dh) from transposed tile: contiguous ---
        v16h vf[4];
#pragma unroll
        for (int t = 0; t < 4; ++t)
        {
            ((uint4*)&vf[t])[0] = *(const uint4*)(Vb + (t * 16 + M) * KT + offA);
            ((uint4*)&vf[t])[1] = *(const uint4*)(Vb + (t * 16 + M) * KT + offA + 16);
        }

        // ---- scores + online softmax per row block ----------------------
#pragma unroll
        for (int rb = 0; rb < 2; ++rb) {
            v8f c0 = {}, c1 = {};
            c0 = __builtin_amdgcn_wmma_f32_16x16x32_f16(false, qa[rb][0].v, false, kf[0][0],
                                                        (short)0, c0, false, false);
            c0 = __builtin_amdgcn_wmma_f32_16x16x32_f16(false, qa[rb][1].v, false, kf[0][1],
                                                        (short)0, c0, false, false);
            c1 = __builtin_amdgcn_wmma_f32_16x16x32_f16(false, qa[rb][0].v, false, kf[1][0],
                                                        (short)0, c1, false, false);
            c1 = __builtin_amdgcn_wmma_f32_16x16x32_f16(false, qa[rb][1].v, false, kf[1][1],
                                                        (short)0, c1, false, false);
            FV sA, sB; sA.v = c0; sB.v = c1;

            float alpha[8];
#pragma unroll
            for (int r = 0; r < 8; ++r) {
                // row max across 16 lanes of this half-wave (+2 tiles)
                float mx = fmaxf(sA.f[r], sB.f[r]);
                mx = fmaxf(mx, __shfl_xor(mx, 1));
                mx = fmaxf(mx, __shfl_xor(mx, 2));
                mx = fmaxf(mx, __shfl_xor(mx, 4));
                mx = fmaxf(mx, __shfl_xor(mx, 8));
                const float mold = mrow[rb][r];
                const float mnew = fmaxf(mold, mx);
                alpha[r]    = exp2f((mold - mnew) * CSCALE);
                mrow[rb][r] = mnew;
                const float p0 = exp2f((sA.f[r] - mnew) * CSCALE);
                const float p1 = exp2f((sB.f[r] - mnew) * CSCALE);
                sA.f[r] = p0; sB.f[r] = p1;
                float rs = p0 + p1;
                rs += __shfl_xor(rs, 1);
                rs += __shfl_xor(rs, 2);
                rs += __shfl_xor(rs, 4);
                rs += __shfl_xor(rs, 8);
                lrow[rb][r] = lrow[rb][r] * alpha[r] + rs;
            }
            // P (C-layout) -> per-wave LDS scratch, f16
#pragma unroll
            for (int r = 0; r < 8; ++r) {
                const int row = rb * 16 + (hi ? 8 + r : r);
                Plds[wv][row * KT + M]      = (_Float16)sA.f[r];
                Plds[wv][row * KT + 16 + M] = (_Float16)sB.f[r];
            }
            // rescale accumulator by alpha
#pragma unroll
            for (int t = 0; t < 4; ++t) {
                FV of; of.v = o[rb][t];
#pragma unroll
                for (int r = 0; r < 8; ++r) of.f[r] *= alpha[r];
                o[rb][t] = of.v;
            }
        }

        // ---- P A-fragments + O += P @ V (K=32 per WMMA) ------------------
#pragma unroll
        for (int rb = 0; rb < 2; ++rb) {
            HV pa;
            const _Float16* prow = &Plds[wv][(rb * 16 + M) * KT];
            *(uint4*)&pa.h[0] = *(const uint4*)(prow + offA);
            *(uint4*)&pa.h[8] = *(const uint4*)(prow + offA + 16);
#pragma unroll
            for (int t = 0; t < 4; ++t)
                o[rb][t] = __builtin_amdgcn_wmma_f32_16x16x32_f16(false, pa.v, false, vf[t],
                                                                 (short)0, o[rb][t], false, false);
        }
        __syncthreads();   // everyone done with buf before it is re-staged
    }

    // ---- epilogue: O * (1/l) -> out[B,S,D] f32 (v_rcp + mul, no divides) -
#pragma unroll
    for (int rb = 0; rb < 2; ++rb) {
        float rinv[8];
#pragma unroll
        for (int r = 0; r < 8; ++r) rinv[r] = __builtin_amdgcn_rcpf(lrow[rb][r]);
#pragma unroll
        for (int t = 0; t < 4; ++t) {
            FV of; of.v = o[rb][t];
            const int n = t * 16 + M;
#pragma unroll
            for (int r = 0; r < 8; ++r) {
                const int row = hi ? 8 + r : r;
                const int s   = q0 + rb * 16 + row;
                out[((size_t)b * S_ + s) * D_ + h * DH_ + n] = of.f[r] * rinv[r];
            }
        }
    }
}

// ---------------------------------------------------------------------------
extern "C" void kernel_launch(void* const* d_in, const int* in_sizes, int n_in,
                              void* d_out, int out_size, void* d_ws, size_t ws_size,
                              hipStream_t stream) {
    const float* x  = (const float*)d_in[0];
    const float* Wq = (const float*)d_in[1];
    const float* bq = (const float*)d_in[2];
    const float* Wk = (const float*)d_in[3];
    const float* bk = (const float*)d_in[4];
    const float* Wv = (const float*)d_in[5];
    const float* bv = (const float*)d_in[6];
    float* out = (float*)d_out;

    const size_t QKV_BYTES = (size_t)B_ * H_ * S_ * DH_ * sizeof(_Float16); // 16 MB
    const size_t X_ELEMS   = (size_t)B_ * S_ * D_;                          // 8.4 M
    const size_t W_ELEMS   = (size_t)H_ * DH_ * DH_;                        // 64 K
    char* ws = (char*)d_ws;
    _Float16* Qw  = (_Float16*)(ws);
    _Float16* Kw  = (_Float16*)(ws + QKV_BYTES);
    _Float16* Vw  = (_Float16*)(ws + 2 * QKV_BYTES);   // transposed [B,H,DH,S]
    _Float16* Xh  = (_Float16*)(ws + 3 * QKV_BYTES);
    _Float16* Wqh = (_Float16*)(ws + 3 * QKV_BYTES + X_ELEMS * 2);
    _Float16* Wkh = Wqh + W_ELEMS;
    _Float16* Wvh = Wkh + W_ELEMS;

    // ---- bulk f32 -> f16 pre-conversion (pure bandwidth, ~33 MB) ---------
    f32_to_f16_kernel<<<X_ELEMS / 2048, 256, 0, stream>>>(x, Xh);
    f32_to_f16_kernel<<<W_ELEMS / 2048, 256, 0, stream>>>(Wq, Wqh);
    f32_to_f16_kernel<<<W_ELEMS / 2048, 256, 0, stream>>>(Wk, Wkh);
    f32_to_f16_kernel<<<W_ELEMS / 2048, 256, 0, stream>>>(Wv, Wvh);

    // 8192 waves total, 8 waves per block
    qkv_proj_kernel<<<(B_ * S_ / 16) * H_ / 8, 256, 0, stream>>>(
        Xh, Wqh, Wkh, Wvh, bq, bk, bv, Qw, Kw, Vw);

    // B*H*(S/QBLK) = 4*16*8 = 512 workgroups
    attn_kernel<<<B_ * H_ * (S_ / QBLK), 256, 0, stream>>>(Qw, Kw, Vw, out);
}
